// MyModel_36189394437070
// MI455X (gfx1250) — compile-verified, matrix operand order checked by reading, not run on hardware
//
#include <hip/hip_runtime.h>
#include <hip/hip_bf16.h>
#include <math.h>

#define NNODES 50000
#define NEDGES 800000
#define NB     256
#define NIN    1024
#define NHID   256

typedef __bf16 bf16_t;
typedef bf16_t v16bf __attribute__((ext_vector_type(16)));
typedef bf16_t v8bf  __attribute__((ext_vector_type(8)));
typedef float  v8f   __attribute__((ext_vector_type(8)));

// ---------------------------------------------------------------------------
// Weight pre-transpose + bf16 convert: WT[n*K + k] = bf16(W[k*N + n])
// ---------------------------------------------------------------------------
__global__ void weight_transpose_kernel(const float* __restrict__ W, bf16_t* __restrict__ WT,
                                        int K, int N) {
    int i = blockIdx.x * blockDim.x + threadIdx.x;
    if (i >= K * N) return;
    int k = i / N, n = i - k * N;
    WT[(size_t)n * K + k] = (bf16_t)W[i];
}

// ---------------------------------------------------------------------------
// Fragment helpers (CDNA5 16-bit 16x32 A/B layouts, wave32)
// lane<16: kb=0, lane>=16: kb=8; elems 0..7 = K kb..kb+7, elems 8..15 = K kb+16..kb+23
// ---------------------------------------------------------------------------
__device__ inline v16bf cvt_a_frag(const float* __restrict__ p) {
    const float4* q = reinterpret_cast<const float4*>(p);
    float4 a0 = q[0];   // kb+0..3
    float4 a1 = q[1];   // kb+4..7
    float4 a2 = q[4];   // kb+16..19
    float4 a3 = q[5];   // kb+20..23
    v16bf r;
    r[0]  = (bf16_t)a0.x; r[1]  = (bf16_t)a0.y; r[2]  = (bf16_t)a0.z; r[3]  = (bf16_t)a0.w;
    r[4]  = (bf16_t)a1.x; r[5]  = (bf16_t)a1.y; r[6]  = (bf16_t)a1.z; r[7]  = (bf16_t)a1.w;
    r[8]  = (bf16_t)a2.x; r[9]  = (bf16_t)a2.y; r[10] = (bf16_t)a2.z; r[11] = (bf16_t)a2.w;
    r[12] = (bf16_t)a3.x; r[13] = (bf16_t)a3.y; r[14] = (bf16_t)a3.z; r[15] = (bf16_t)a3.w;
    return r;
}

__device__ inline v16bf load_b_frag(const bf16_t* __restrict__ p) {
    v8bf lo = *reinterpret_cast<const v8bf*>(p);        // kb..kb+7
    v8bf hi = *reinterpret_cast<const v8bf*>(p + 16);   // kb+16..kb+23
    return __builtin_shufflevector(lo, hi, 0,1,2,3,4,5,6,7,8,9,10,11,12,13,14,15);
}

// ---------------------------------------------------------------------------
// WMMA bf16 GEMM: C[M,N] = act(rowscale(row) * (A @ W) + bias)
// A: [M,K] f32 row-major; WT: [N,K] bf16 (pre-transposed W).
// Block = 256 threads (8 waves); each wave owns a 16x16 tile; block covers
// 16 rows x 128 cols. grid.x = N/128, grid.y = M/16. No LDS, no barriers.
// scale_mode: 0 = none, 1 = *rowscale, 2 = *(1-rowscale)
// ---------------------------------------------------------------------------
__global__ __launch_bounds__(256)
void gemm_wmma_kernel(const float* __restrict__ A, const bf16_t* __restrict__ WT,
                      const float* __restrict__ bias, const float* __restrict__ rowscale,
                      float* __restrict__ C, int M, int K, int N,
                      int scale_mode, int relu)
{
    const int lane = threadIdx.x & 31;
    const int wave = threadIdx.x >> 5;
    const int rowbase = blockIdx.y * 16;
    const int colbase = blockIdx.x * 128;

    const int m  = lane & 15;
    const int kb = (lane < 16) ? 0 : 8;
    int grow = rowbase + m;
    if (grow >= M) grow = M - 1;                       // M % 16 == 0 here anyway
    const float*  arow = A  + (size_t)grow * K + kb;
    const int ncol = colbase + wave * 16 + (lane & 15);
    const bf16_t* wrow = WT + (size_t)ncol * K + kb;

    v8f acc0 = {}, acc1 = {};
    for (int k0 = 0; k0 < K; k0 += 64) {
        if (k0 + 64 < K) __builtin_prefetch(arow + k0 + 64, 0, 1);
        v16bf a0 = cvt_a_frag(arow + k0);
        v16bf b0 = load_b_frag(wrow + k0);
        acc0 = __builtin_amdgcn_wmma_f32_16x16x32_bf16(false, a0, false, b0,
                                                       (short)0, acc0, false, false);
        v16bf a1 = cvt_a_frag(arow + k0 + 32);
        v16bf b1 = load_b_frag(wrow + k0 + 32);
        acc1 = __builtin_amdgcn_wmma_f32_16x16x32_bf16(false, a1, false, b1,
                                                       (short)0, acc1, false, false);
    }

    // Epilogue: C/D layout — VGPR r: lanes 0-15 -> M=r, lanes 16-31 -> M=r+8; N=lane&15
    const int col = colbase + wave * 16 + (lane & 15);
    #pragma unroll
    for (int r = 0; r < 8; ++r) {
        int row = rowbase + r + ((lane < 16) ? 0 : 8);
        if (row < M && col < N) {
            float v = acc0[r] + acc1[r];
            if (scale_mode) {
                float s = rowscale[row];
                if (scale_mode == 2) s = 1.0f - s;
                v *= s;
            }
            if (bias) v += bias[col];
            if (relu) v = fmaxf(v, 0.0f);
            C[(size_t)row * N + col] = v;
        }
    }
}

// ---------------------------------------------------------------------------
// GCN normalization helpers
// ---------------------------------------------------------------------------
__global__ void deg_init_kernel(float* deg, int n) {
    int i = blockIdx.x * blockDim.x + threadIdx.x;
    if (i < n) deg[i] = 1.0f;  // self loop weight
}

__global__ void deg_accum_kernel(float* deg, const int* __restrict__ col,
                                 const float* __restrict__ esc, int mode, int e) {
    int i = blockIdx.x * blockDim.x + threadIdx.x;
    if (i >= e) return;
    float w = 1.0f;
    if (mode == 1) w = esc[i];
    else if (mode == 2) w = 1.0f - esc[i];
    atomicAdd(&deg[col[i]], w);
}

__global__ void deg_rsqrt_kernel(float* deg, int n) {
    int i = blockIdx.x * blockDim.x + threadIdx.x;
    if (i < n) deg[i] = rsqrtf(fmaxf(deg[i], 1e-12f));   // deg >= 1 always
}

// out[n,h] = xw[n,h] * dis[n]^2 + bias[h]   (self-loop term + conv bias)
__global__ void agg_init_kernel(float* __restrict__ out, const float* __restrict__ xw,
                                const float* __restrict__ dis, const float* __restrict__ bias,
                                int total) {
    int i = blockIdx.x * blockDim.x + threadIdx.x;
    if (i >= total) return;
    int n = i >> 8;                 // / NHID
    int h = i & 255;
    float d = dis[n];
    out[i] = xw[i] * d * d + bias[h];
}

// one wave per edge: out[col] += xw[row] * dis[row]*ew*dis[col]
__global__ __launch_bounds__(256)
void scatter_kernel(float* __restrict__ out, const float* __restrict__ xw,
                    const float* __restrict__ dis,
                    const int* __restrict__ row, const int* __restrict__ col,
                    const float* __restrict__ esc, int mode, int e) {
    int gid  = blockIdx.x * blockDim.x + threadIdx.x;
    int edge = gid >> 5;
    int lane = gid & 31;
    if (edge >= e) return;
    int r = row[edge], c = col[edge];
    float w = 1.0f;
    if (mode == 1) w = esc[edge];
    else if (mode == 2) w = 1.0f - esc[edge];
    w *= dis[r] * dis[c];
    const float* src = xw + (size_t)r * NHID;
    float* dst = out + (size_t)c * NHID;
    #pragma unroll
    for (int i = 0; i < 8; ++i) {
        int h = lane + i * 32;
        atomicAdd(&dst[h], src[h] * w);
    }
}

__global__ void relu_kernel(float* a, int total) {
    int i = blockIdx.x * blockDim.x + threadIdx.x;
    if (i < total) a[i] = fmaxf(a[i], 0.0f);
}

__global__ void zero_kernel(float* a, int total) {
    int i = blockIdx.x * blockDim.x + threadIdx.x;
    if (i < total) a[i] = 0.0f;
}

// ---------------------------------------------------------------------------
// Masker node/edge scores: one wave per node, 3 dot products over H=256
// ---------------------------------------------------------------------------
__global__ __launch_bounds__(256)
void nodescore_kernel(const float* __restrict__ h, const float* __restrict__ node_w,
                      const float* __restrict__ node_b, const float* __restrict__ edge_w,
                      float* __restrict__ ns, float* __restrict__ e1, float* __restrict__ e2,
                      int n_nodes) {
    int wave = threadIdx.x >> 5, lane = threadIdx.x & 31;
    int node = blockIdx.x * 8 + wave;
    if (node >= n_nodes) return;
    const float* hp = h + (size_t)node * NHID;
    float s0 = 0.f, s1 = 0.f, s2 = 0.f;
    #pragma unroll
    for (int i = 0; i < 8; ++i) {
        int idx = lane + i * 32;
        float v = hp[idx];
        s0 += v * node_w[idx];
        s1 += v * edge_w[idx];
        s2 += v * edge_w[NHID + idx];
    }
    #pragma unroll
    for (int off = 16; off > 0; off >>= 1) {
        s0 += __shfl_xor(s0, off, 32);
        s1 += __shfl_xor(s1, off, 32);
        s2 += __shfl_xor(s2, off, 32);
    }
    if (lane == 0) {
        ns[node] = 1.0f / (1.0f + expf(-(s0 + node_b[0])));
        e1[node] = s1;
        e2[node] = s2;
    }
}

__global__ void edgescore_kernel(const float* __restrict__ e1, const float* __restrict__ e2,
                                 const int* __restrict__ row, const int* __restrict__ col,
                                 const float* __restrict__ eb, float* __restrict__ esc, int e) {
    int i = blockIdx.x * blockDim.x + threadIdx.x;
    if (i >= e) return;
    float s = e1[row[i]] + e2[col[i]] + eb[0];
    esc[i] = 1.0f / (1.0f + expf(-s));
}

// ---------------------------------------------------------------------------
// Segment-mean pooling
// ---------------------------------------------------------------------------
__global__ void pool_accum_kernel(const float* __restrict__ h, const int* __restrict__ batch,
                                  float* __restrict__ score, int total) {
    int i = blockIdx.x * blockDim.x + threadIdx.x;
    if (i >= total) return;
    int n = i >> 8, hh = i & 255;
    atomicAdd(&score[(size_t)batch[n] * NHID + hh], h[i]);
}

__global__ void cnt_accum_kernel(const int* __restrict__ batch, float* __restrict__ cnt, int n) {
    int i = blockIdx.x * blockDim.x + threadIdx.x;
    if (i < n) atomicAdd(&cnt[batch[i]], 1.0f);
}

__global__ void pool_div_kernel(float* score, const float* __restrict__ cnt, int total) {
    int i = blockIdx.x * blockDim.x + threadIdx.x;
    if (i >= total) return;
    score[i] /= fmaxf(cnt[i >> 8], 1.0f);
}

// ---------------------------------------------------------------------------
// Heads + CE/GCE losses (single block, one thread per graph)
// ---------------------------------------------------------------------------
__device__ inline float lse2(float a, float b) {
    float m = fmaxf(a, b);
    return m + logf(expf(a - m) + expf(b - m));
}
__device__ inline float cef(float l0, float l1, int y) {
    float l = lse2(l0, l1);
    return l - (y ? l1 : l0);
}
__device__ inline float gcef(float l0, float l1, int y) {
    float l  = lse2(l0, l1);
    float ly = y ? l1 : l0;
    float py = expf(ly - l);
    return (l - ly) * powf(py, 0.7f) * 0.7f;
}

__global__ __launch_bounds__(NB)
void head_loss_kernel(const float* __restrict__ sc, const float* __restrict__ sb,
                      const float* __restrict__ wc, const float* __restrict__ bc,
                      const float* __restrict__ wb, const float* __restrict__ bb,
                      const int* __restrict__ y, const int* __restrict__ perm,
                      float* __restrict__ out) {
    __shared__ float red[NB];
    int b = threadIdx.x;
    const float* c  = sc + (size_t)b * NHID;
    const float* bv = sb + (size_t)b * NHID;
    const float* bs = sb + (size_t)perm[b] * NHID;

    float pc0 = bc[0], pc1 = bc[1], pb0 = bb[0], pb1 = bb[1];
    float mc0 = bc[0], mc1 = bc[1], ma0 = bb[0], ma1 = bb[1];
    for (int k = 0; k < NHID; ++k) {
        float vc = c[k];
        float wc0 = wc[k * 2], wc1 = wc[k * 2 + 1];
        float wb0 = wb[k * 2], wb1 = wb[k * 2 + 1];
        pc0 += vc * wc0; pc1 += vc * wc1;
        pb0 += vc * wb0; pb1 += vc * wb1;
        mc0 += vc * wc0; mc1 += vc * wc1;
        ma0 += vc * wb0; ma1 += vc * wb1;
    }
    for (int k = 0; k < NHID; ++k) {
        float vb = bv[k], vs = bs[k];
        float wc0 = wc[(NHID + k) * 2], wc1 = wc[(NHID + k) * 2 + 1];
        float wb0 = wb[(NHID + k) * 2], wb1 = wb[(NHID + k) * 2 + 1];
        pc0 += vb * wc0; pc1 += vb * wc1;
        pb0 += vb * wb0; pb1 += vb * wb1;
        mc0 += vs * wc0; mc1 += vs * wc1;
        ma0 += vs * wb0; ma1 += vs * wb1;
    }
    int yy = y[b], ys = y[perm[b]];
    out[b * 2 + 0] = pc0;
    out[b * 2 + 1] = pc1;
    float t = cef(pc0, pc1, yy) + gcef(pb0, pb1, yy)
            + 15.0f * (cef(mc0, mc1, yy) + gcef(ma0, ma1, ys));
    red[b] = t;
    __syncthreads();
    for (int s = NB / 2; s > 0; s >>= 1) {
        if (b < s) red[b] += red[b + s];
        __syncthreads();
    }
    if (b == 0) out[NB * 2] = red[0] / (float)NB;
}

// ---------------------------------------------------------------------------
// Host orchestration
// ---------------------------------------------------------------------------
extern "C" void kernel_launch(void* const* d_in, const int* in_sizes, int n_in,
                              void* d_out, int out_size, void* d_ws, size_t ws_size,
                              hipStream_t stream) {
    const float* x     = (const float*)d_in[0];
    const int*   ei    = (const int*)d_in[1];
    const int*   batch = (const int*)d_in[2];
    const int*   y     = (const int*)d_in[3];
    const int*   perm  = (const int*)d_in[4];
    const float* m_fc_w = (const float*)d_in[5];  const float* m_fc_b = (const float*)d_in[6];
    const float* m_c1_w = (const float*)d_in[7];  const float* m_c1_b = (const float*)d_in[8];
    const float* m_c2_w = (const float*)d_in[9];  const float* m_c2_b = (const float*)d_in[10];
    const float* m_node_w = (const float*)d_in[11]; const float* m_node_b = (const float*)d_in[12];
    const float* m_edge_w = (const float*)d_in[13]; const float* m_edge_b = (const float*)d_in[14];
    const float* c_fc_w = (const float*)d_in[15]; const float* c_fc_b = (const float*)d_in[16];
    const float* c_c1_w = (const float*)d_in[17]; const float* c_c1_b = (const float*)d_in[18];
    const float* c_c2_w = (const float*)d_in[19]; const float* c_c2_b = (const float*)d_in[20];
    const float* b_fc_w = (const float*)d_in[21]; const float* b_fc_b = (const float*)d_in[22];
    const float* b_c1_w = (const float*)d_in[23]; const float* b_c1_b = (const float*)d_in[24];
    const float* b_c2_w = (const float*)d_in[25]; const float* b_c2_b = (const float*)d_in[26];
    const float* mlp_c_w = (const float*)d_in[27]; const float* mlp_c_b = (const float*)d_in[28];
    const float* mlp_b_w = (const float*)d_in[29]; const float* mlp_b_b = (const float*)d_in[30];

    const int* row = ei;
    const int* col = ei + NEDGES;

    // workspace carve
    char* wsp = (char*)d_ws;
    auto alloc = [&](size_t bytes) {
        void* p = (void*)wsp;
        wsp += (bytes + 255) & ~(size_t)255;
        return p;
    };
    const size_t NHbytes = (size_t)NNODES * NHID * sizeof(float);
    float* bufA  = (float*)alloc(NHbytes);
    float* bufB  = (float*)alloc(NHbytes);
    float* bufC  = (float*)alloc(NHbytes);
    float* ns    = (float*)alloc((size_t)NNODES * sizeof(float));
    float* e1    = (float*)alloc((size_t)NNODES * sizeof(float));
    float* e2    = (float*)alloc((size_t)NNODES * sizeof(float));
    float* esc   = (float*)alloc((size_t)NEDGES * sizeof(float));
    float* dism  = (float*)alloc((size_t)NNODES * sizeof(float));
    float* disc  = (float*)alloc((size_t)NNODES * sizeof(float));
    float* disb  = (float*)alloc((size_t)NNODES * sizeof(float));
    float* score_c = (float*)alloc((size_t)NB * NHID * sizeof(float));
    float* score_b = (float*)alloc((size_t)NB * NHID * sizeof(float));
    float* cnt   = (float*)alloc((size_t)NB * sizeof(float));
    // bf16 transposed weights
    bf16_t* wt_m_fc = (bf16_t*)alloc((size_t)NIN  * NHID * sizeof(bf16_t));
    bf16_t* wt_m_c1 = (bf16_t*)alloc((size_t)NHID * NHID * sizeof(bf16_t));
    bf16_t* wt_m_c2 = (bf16_t*)alloc((size_t)NHID * NHID * sizeof(bf16_t));
    bf16_t* wt_c_fc = (bf16_t*)alloc((size_t)NIN  * NHID * sizeof(bf16_t));
    bf16_t* wt_c_c1 = (bf16_t*)alloc((size_t)NHID * NHID * sizeof(bf16_t));
    bf16_t* wt_c_c2 = (bf16_t*)alloc((size_t)NHID * NHID * sizeof(bf16_t));
    bf16_t* wt_b_fc = (bf16_t*)alloc((size_t)NIN  * NHID * sizeof(bf16_t));
    bf16_t* wt_b_c1 = (bf16_t*)alloc((size_t)NHID * NHID * sizeof(bf16_t));
    bf16_t* wt_b_c2 = (bf16_t*)alloc((size_t)NHID * NHID * sizeof(bf16_t));

    const int NHtot = NNODES * NHID;       // 12.8M
    const dim3 gemm_grid(NHID / 128, (NNODES + 15) / 16);
    const int TB = 256;
    const int gN  = (NNODES + TB - 1) / TB;
    const int gE  = (NEDGES + TB - 1) / TB;
    const int gNH = (NHtot + TB - 1) / TB;
    const int gEw = (NEDGES * 32 + TB - 1) / TB;   // wave per edge

    // ---- weight transposes (bf16) ----
    #define WT(W, DST, K) weight_transpose_kernel<<<((K)*NHID + TB - 1) / TB, TB, 0, stream>>>(W, DST, K, NHID)
    WT(m_fc_w, wt_m_fc, NIN);  WT(m_c1_w, wt_m_c1, NHID);  WT(m_c2_w, wt_m_c2, NHID);
    WT(c_fc_w, wt_c_fc, NIN);  WT(c_c1_w, wt_c_c1, NHID);  WT(c_c2_w, wt_c_c2, NHID);
    WT(b_fc_w, wt_b_fc, NIN);  WT(b_c1_w, wt_b_c1, NHID);  WT(b_c2_w, wt_b_c2, NHID);
    #undef WT

    #define GEMM(A, WTB, BIAS, SCALE, C, K, MODE, RELU) \
        gemm_wmma_kernel<<<gemm_grid, TB, 0, stream>>>(A, WTB, BIAS, SCALE, C, NNODES, K, NHID, MODE, RELU)
    #define CONV(XIN, WTB, BVEC, HOUT, TMP, DIS, EMODE)                                   \
        do {                                                                              \
            GEMM(XIN, WTB, nullptr, nullptr, TMP, NHID, 0, 0);                            \
            agg_init_kernel<<<gNH, TB, 0, stream>>>(HOUT, TMP, DIS, BVEC, NHtot);         \
            scatter_kernel<<<gEw, TB, 0, stream>>>(HOUT, TMP, DIS, row, col, esc, EMODE, NEDGES); \
            relu_kernel<<<gNH, TB, 0, stream>>>(HOUT, NHtot);                             \
        } while (0)

    // ---- unweighted degree norm for masker ----
    deg_init_kernel<<<gN, TB, 0, stream>>>(dism, NNODES);
    deg_accum_kernel<<<gE, TB, 0, stream>>>(dism, col, esc, 0, NEDGES);
    deg_rsqrt_kernel<<<gN, TB, 0, stream>>>(dism, NNODES);

    // ---- GCNMasker ----
    GEMM(x, wt_m_fc, m_fc_b, nullptr, bufA, NIN, 0, 0);      // h0
    CONV(bufA, wt_m_c1, m_c1_b, bufC, bufB, dism, 0);        // h1 -> bufC
    CONV(bufC, wt_m_c2, m_c2_b, bufB, bufA, dism, 0);        // h2 -> bufB
    nodescore_kernel<<<(NNODES + 7) / 8, TB, 0, stream>>>(bufB, m_node_w, m_node_b,
                                                          m_edge_w, ns, e1, e2, NNODES);
    edgescore_kernel<<<gE, TB, 0, stream>>>(e1, e2, row, col, m_edge_b, esc, NEDGES);

    // ---- weighted degree norms ----
    deg_init_kernel<<<gN, TB, 0, stream>>>(disc, NNODES);
    deg_accum_kernel<<<gE, TB, 0, stream>>>(disc, col, esc, 1, NEDGES);
    deg_rsqrt_kernel<<<gN, TB, 0, stream>>>(disc, NNODES);
    deg_init_kernel<<<gN, TB, 0, stream>>>(disb, NNODES);
    deg_accum_kernel<<<gE, TB, 0, stream>>>(disb, col, esc, 2, NEDGES);
    deg_rsqrt_kernel<<<gN, TB, 0, stream>>>(disb, NNODES);

    // ---- gnn_c: h = ns * (x @ c_fc_w) + b ----
    GEMM(x, wt_c_fc, c_fc_b, ns, bufA, NIN, 1, 0);
    CONV(bufA, wt_c_c1, c_c1_b, bufC, bufB, disc, 1);
    CONV(bufC, wt_c_c2, c_c2_b, bufB, bufA, disc, 1);
    zero_kernel<<<(NB * NHID + TB - 1) / TB, TB, 0, stream>>>(score_c, NB * NHID);
    zero_kernel<<<1, TB, 0, stream>>>(cnt, NB);
    pool_accum_kernel<<<gNH, TB, 0, stream>>>(bufB, batch, score_c, NHtot);
    cnt_accum_kernel<<<gN, TB, 0, stream>>>(batch, cnt, NNODES);
    pool_div_kernel<<<(NB * NHID + TB - 1) / TB, TB, 0, stream>>>(score_c, cnt, NB * NHID);

    // ---- gnn_b: h = (1-ns) * (x @ b_fc_w) + b ----
    GEMM(x, wt_b_fc, b_fc_b, ns, bufA, NIN, 2, 0);
    CONV(bufA, wt_b_c1, b_c1_b, bufC, bufB, disb, 2);
    CONV(bufC, wt_b_c2, b_c2_b, bufB, bufA, disb, 2);
    zero_kernel<<<(NB * NHID + TB - 1) / TB, TB, 0, stream>>>(score_b, NB * NHID);
    pool_accum_kernel<<<gNH, TB, 0, stream>>>(bufB, batch, score_b, NHtot);
    pool_div_kernel<<<(NB * NHID + TB - 1) / TB, TB, 0, stream>>>(score_b, cnt, NB * NHID);

    // ---- heads + loss ----
    head_loss_kernel<<<1, NB, 0, stream>>>(score_c, score_b, mlp_c_w, mlp_c_b,
                                           mlp_b_w, mlp_b_b, y, perm, (float*)d_out);
    #undef CONV
    #undef GEMM
}
